// compression_module_12197707121030
// MI455X (gfx1250) — compile-verified, matrix operand order checked by reading, not run on hardware
//
#include <hip/hip_runtime.h>
#include <hip/hip_bf16.h>

// ---------------------------------------------------------------------------
// CDNA5 / gfx1250 implementation: conv3x3 -> BN -> sigmoid -> BEC ->
// deconv(k2,s2) -> BN -> relu, all GEMMs via v_wmma_f32_16x16x32_bf16.
// BN batch statistics fused into GEMM epilogues (shfl reduce + f32 atomics).
// Data movement: per-lane global_load_async_to_lds_b128 for streamed tiles,
// Tensor Data Mover (tensor_load_to_lds) for the deconv 128KB weight preload.
// Workspace usage: ~52.3 MB (z1/h buffer + stats + bf16 weights).
// ---------------------------------------------------------------------------

typedef unsigned int  uint;
typedef unsigned short ushort;
typedef __attribute__((ext_vector_type(16))) __bf16 v16bf;
typedef __attribute__((ext_vector_type(8)))  float  v8f;
typedef int v4i __attribute__((vector_size(16)));   // matches async builtin param
typedef unsigned int u32x4 __attribute__((vector_size(16)));
typedef int i32x8 __attribute__((vector_size(32)));
typedef int i32x4 __attribute__((vector_size(16)));

// ---- optional CDNA5 async global->LDS path (guarded; fallback always OK) ---
#if defined(__has_builtin)
#  if __has_builtin(__builtin_amdgcn_global_load_async_to_lds_b128) && \
      __has_builtin(__builtin_amdgcn_s_wait_asynccnt)
#    define ATHENA_ASYNC_LDS 1
#  endif
#  if __has_builtin(__builtin_amdgcn_tensor_load_to_lds) && \
      __has_builtin(__builtin_amdgcn_s_wait_tensorcnt)
#    define ATHENA_TDM 1
#  endif
#endif
#ifndef ATHENA_ASYNC_LDS
#  define ATHENA_ASYNC_LDS 0
#endif
#ifndef ATHENA_TDM
#  define ATHENA_TDM 0
#endif

#if ATHENA_ASYNC_LDS
__device__ __forceinline__ void async_g2l_b128(const void* g, void* lds) {
  __builtin_amdgcn_global_load_async_to_lds_b128(
      (__attribute__((address_space(1))) v4i*)(unsigned long long)(g),
      (__attribute__((address_space(3))) v4i*)(unsigned)(unsigned long long)(lds),
      0, 0);
}
#endif

// float -> bf16, round-to-nearest-even
__device__ __forceinline__ ushort f2bf(float f) {
  uint u = __float_as_uint(f);
  u += 0x7fffu + ((u >> 16) & 1u);
  return (ushort)(u >> 16);
}

union BFPack { uint u[8]; v16bf v; };

__device__ __forceinline__ uint hash32(uint x) {
  x ^= x >> 16; x *= 0x7feb352du;
  x ^= x >> 15; x *= 0x846ca68bu;
  x ^= x >> 16; return x;
}

// Problem sizes
#define B_    32
#define C_    256
#define H_    56
#define HW_   3136          // 56*56
#define HC_   128
#define K1    2313          // 257*9
#define K1P   2336          // padded to 73*32
#define NTOT  100352        // 32*56*56
#define Z1_ELEMS   12845056 // 32*128*56*56
#define OUT_ELEMS  102760448L
#define W1BF_ELEMS (128 * 2336)
#define W4BF_ELEMS (4 * 256 * 128)

// ---------------------------------------------------------------------------
__global__ __launch_bounds__(256) void zero_stats(float* s) {
  for (int i = threadIdx.x; i < 1536; i += 256) s[i] = 0.0f;
}

// Pre-convert weights to bf16. w1: [128][2313] -> [128][2336] zero-padded.
// w4 (ConvTranspose [ci][co][ky][kx]) -> w4bf[tap][co][ci].
__global__ __launch_bounds__(256) void cvt_weights(
    const float* __restrict__ w1, const float* __restrict__ w4,
    ushort* __restrict__ w1bf, ushort* __restrict__ w4bf) {
  int i = blockIdx.x * 256 + threadIdx.x;
  if (i < W1BF_ELEMS) {
    int oc = i / K1P, k = i - oc * K1P;
    float v = (k < K1) ? w1[oc * K1 + k] : 0.0f;
    w1bf[i] = f2bf(v);
  }
  if (i < W4BF_ELEMS) {
    int t = i >> 15, rem = i & 32767;
    int co = rem >> 7, ci = rem & 127;
    w4bf[i] = f2bf(w4[ci * 1024 + co * 4 + t]);
  }
}

// ---------------------------------------------------------------------------
// Conv1 implicit GEMM: z1[oc, n] = sum_k W[oc,k] * X[k,n]  (+b1)
// Block: 256 thr (8 waves). Each block: 128 x 16 output tile; wave w owns
// rows [16w,16w+16). K loop in steps of 32 with bf16 WMMA.
__global__ __launch_bounds__(256) void conv1_wmma(
    const float* __restrict__ x, const float* __restrict__ noise_u,
    const ushort* __restrict__ w1bf, const float* __restrict__ b1,
    float* __restrict__ z1, float* __restrict__ sum1, float* __restrict__ sq1) {
  __shared__ alignas(16) ushort Ash[128 * 32];  // weights tile
  __shared__ alignas(16) ushort Bsh[16 * 32];   // input tile, [col][k]
  const int tid = threadIdx.x;
  const int wv = tid >> 5, lane = tid & 31;
  const int hi = lane >> 4, mn = lane & 15;
  const int n0 = blockIdx.x * 16;
  const float nf = noise_u[0] * 0.5f;

  v8f acc = {};
  for (int kb = 0; kb < K1P; kb += 32) {
    // ---- stage A: 128 rows x 32 k of bf16 weights (contiguous per row) ----
#if ATHENA_ASYNC_LDS
    for (int q = tid; q < 512; q += 256) {        // 512 x b128 chunks
      int row = q >> 2, part = q & 3;
      async_g2l_b128(w1bf + row * K1P + kb + part * 8, &Ash[row * 32 + part * 8]);
    }
#else
    for (int q = tid; q < 4096; q += 256) {
      int row = q >> 5, kk = q & 31;
      Ash[q] = w1bf[row * K1P + kb + kk];
    }
#endif
    // ---- stage B: gather 32 k x 16 n input values (3x3 taps, pad=1) ------
    for (int q = tid; q < 512; q += 256) {
      int col = q >> 5, kk = q & 31;
      int n = n0 + col;
      int b = n / HW_, pos = n - b * HW_;
      int yy = pos / H_, xx = pos - yy * H_;
      int k = kb + kk;
      float v = 0.0f;
      if (k < K1) {
        int ci = k / 9, tap = k - ci * 9;
        int ky = tap / 3, kx = tap - ky * 3;
        int iy = yy + ky - 1, ix = xx + kx - 1;
        if ((uint)iy < (uint)H_ && (uint)ix < (uint)H_)
          v = (ci == C_) ? nf : x[((b * C_ + ci) * H_ + iy) * H_ + ix];
      }
      Bsh[col * 32 + kk] = f2bf(v);
    }
#if ATHENA_ASYNC_LDS
    __builtin_amdgcn_s_wait_asynccnt(0);
#endif
    __syncthreads();

    // ---- per-lane gather into WMMA register layout, then matrix op -------
    BFPack a, bm;
    const ushort* aRow = &Ash[(wv * 16 + mn) * 32];
    const ushort* bRow = &Bsh[mn * 32];
#pragma unroll
    for (int v = 0; v < 8; ++v) {
      int k0 = 2 * v + ((v < 4) ? 0 : 8) + 8 * hi;  // paired-K layout
      a.u[v]  = *(const uint*)(aRow + k0);
      bm.u[v] = *(const uint*)(bRow + k0);
    }
    acc = __builtin_amdgcn_wmma_f32_16x16x32_bf16(false, a.v, false, bm.v,
                                                  (short)0, acc, false, false);
    __syncthreads();
  }

  // ---- epilogue: +bias, store z1, fused BN1 partial stats ----------------
  const int n = n0 + mn;
  const int b = n / HW_, pos = n - b * HW_;
  const int yy = pos / H_, xx = pos - yy * H_;
#pragma unroll
  for (int r = 0; r < 8; ++r) {
    int oc = wv * 16 + r + 8 * hi;
    float zv = acc[r] + b1[oc];
    z1[((b * HC_ + oc) * H_ + yy) * H_ + xx] = zv;
    float s = zv, q = zv * zv;
#pragma unroll
    for (int off = 1; off < 16; off <<= 1) {
      s += __shfl_xor(s, off, 32);
      q += __shfl_xor(q, off, 32);
    }
    if (mn == 0) { atomicAdd(&sum1[oc], s); atomicAdd(&sq1[oc], q); }
  }
}

// ---------------------------------------------------------------------------
__global__ void bn_finalize(const float* __restrict__ sum, const float* __restrict__ sq,
                            const float* __restrict__ g, const float* __restrict__ beta,
                            float* __restrict__ a, float* __restrict__ sh,
                            int C, float invN) {
  int c = blockIdx.x * blockDim.x + threadIdx.x;
  if (c < C) {
    float m = sum[c] * invN;
    float v = fmaxf(sq[c] * invN - m * m, 0.0f);
    float s = g[c] * rsqrtf(v + 1e-5f);
    a[c] = s;
    sh[c] = beta[c] - m * s;
  }
}

// BN1 + sigmoid + binary-erasure channel (deterministic hash RNG), in place.
__global__ __launch_bounds__(256) void bn_sig_bec(
    float* __restrict__ z, const float* __restrict__ a, const float* __restrict__ sh,
    const float* __restrict__ noise_u, int total) {
  int i = blockIdx.x * 256 + threadIdx.x;
  if (i >= total) return;
  int c = (i / HW_) & (HC_ - 1);
  float zz = a[c] * z[i] + sh[c];
  float s = 1.0f / (1.0f + __expf(-zz));
  float p = noise_u[0] * 0.5f;
  uint xb = ((uint)(int)rintf(s * 256.0f)) & 255u;   // byte quantize, wrap
  uint thr = (uint)rintf((1.0f - p) * 256.0f);       // P(bit kept) = 1-p
  uint h0 = hash32(0x42000001u ^ (uint)i);
  uint h1 = hash32(0x13371337u + (uint)i * 0x9E3779B9u);
  uint mask = 0;
#pragma unroll
  for (int j = 0; j < 8; ++j) {
    uint byte = ((j < 4) ? (h0 >> (8 * j)) : (h1 >> (8 * (j - 4)))) & 255u;
    mask |= (byte < thr ? 1u : 0u) << j;
  }
  float fm = (float)mask;
  z[i] = ((float)(xb & mask) + (255.0f - fm) * 0.5f) * (1.0f / 255.0f);
}

// ---------------------------------------------------------------------------
// Deconv (k=2,s=2) as 4 tap-GEMMs: y[b,co,2iy+ky,2ix+kx] = b4 + sum_ci
// h[b,ci,iy,ix]*W[tap][co][ci].  grid=(6272 n-tiles, 2 co-halves).
// All 4x128x128 tap weights preloaded once into 128 KB dynamic LDS:
// one TDM descriptor (2D tile: 4 slabs x 16384 bf16, row stride 32768 elems).
__global__ __launch_bounds__(256) void deconv_wmma(
    const float* __restrict__ h, const ushort* __restrict__ w4bf,
    const float* __restrict__ b4, float* __restrict__ y,
    float* __restrict__ sum2, float* __restrict__ sq2) {
  extern __shared__ ushort dsm[];
  ushort* Wsh = dsm;                 // [4][128 co][128 ci]
  ushort* Bsh = dsm + 4 * 128 * 128; // [16 col][32 k]
  const int tid = threadIdx.x;
  const int wv = tid >> 5, lane = tid & 31;
  const int hi = lane >> 4, mn = lane & 15;
  const int n0 = blockIdx.x * 16;
  const int cob = blockIdx.y * 128;

  // ---- preload all tap weights for this co-half ---------------------------
#if ATHENA_TDM
  if (tid == 0) {   // single wave issues the DMA; TDM ignores EXEC
    unsigned long long gaddr = (unsigned long long)(w4bf + (unsigned)cob * 128u);
    uint lds = (uint)(unsigned long long)(&Wsh[0]);
    u32x4 g0 = { 1u,                                  // count=1, user descriptor
                 lds,                                 // lds_addr
                 (uint)gaddr,                         // global_addr[31:0]
                 (uint)(gaddr >> 32) | (2u << 30) };  // global_addr[56:32] | type=2
    i32x8 g1 = { (int)(1u << 16),       // data_size = 2 bytes; mask=0; no flags
                 (int)(0x8000u << 16),  // tensor_dim0 = 32768 (low 16 -> [31:16])
                 (int)(4u << 16),       // tensor_dim0 hi = 0; tensor_dim1 = 4
                 (int)(0x4000u << 16),  // tensor_dim1 hi = 0; tile_dim0 = 16384
                 4,                     // tile_dim1 = 4; tile_dim2 = 0
                 32768,                 // tensor_dim0_stride (low 32)
                 0,                     // stride0 hi; tensor_dim1_stride low
                 0 };
    i32x4 gz4 = { 0, 0, 0, 0 };
    i32x8 gz8 = { 0, 0, 0, 0, 0, 0, 0, 0 };
    __builtin_amdgcn_tensor_load_to_lds(g0, g1, gz4, gz4, gz8, 0);
  }
  __builtin_amdgcn_s_wait_tensorcnt(0);
#elif ATHENA_ASYNC_LDS
  for (int q = tid; q < 8192; q += 256) {          // 8192 x b128 chunks
    int row = q >> 4, part = q & 15;
    int t = row >> 7, col = row & 127;
    async_g2l_b128(w4bf + (t * 256 + cob + col) * 128 + part * 8,
                   &Wsh[row * 128 + part * 8]);
  }
  __builtin_amdgcn_s_wait_asynccnt(0);
#else
  for (int q = tid; q < 65536; q += 256) {
    int row = q >> 7, ci = q & 127;
    int t = row >> 7, col = row & 127;
    Wsh[q] = w4bf[(t * 256 + cob + col) * 128 + ci];
  }
#endif
  __syncthreads();

  const int nn = n0 + mn;
  const int bb = nn / HW_, pos = nn - bb * HW_;
  const int iy = pos / H_, ix = pos - iy * H_;

  v8f acc[4];
#pragma unroll
  for (int t = 0; t < 4; ++t)
#pragma unroll
    for (int r = 0; r < 8; ++r)
      acc[t][r] = b4[cob + wv * 16 + r + 8 * hi];

  for (int kb = 0; kb < 128; kb += 32) {
    __syncthreads();
    for (int q = tid; q < 512; q += 256) {
      int col = q >> 5, kk = q & 31;
      int n = n0 + col;
      int b = n / HW_, p2 = n - b * HW_;
      int yy = p2 / H_, xx = p2 - yy * H_;
      Bsh[col * 32 + kk] = f2bf(h[((b * HC_ + kb + kk) * H_ + yy) * H_ + xx]);
    }
    __syncthreads();

    BFPack bm;
    const ushort* bRow = &Bsh[mn * 32];
#pragma unroll
    for (int v = 0; v < 8; ++v) {
      int k0 = 2 * v + ((v < 4) ? 0 : 8) + 8 * hi;
      bm.u[v] = *(const uint*)(bRow + k0);
    }
#pragma unroll
    for (int t = 0; t < 4; ++t) {
      BFPack a;
      const ushort* aRow = &Wsh[(t * 128 + wv * 16 + mn) * 128 + kb];
#pragma unroll
      for (int v = 0; v < 8; ++v) {
        int k0 = 2 * v + ((v < 4) ? 0 : 8) + 8 * hi;
        a.u[v] = *(const uint*)(aRow + k0);
      }
      acc[t] = __builtin_amdgcn_wmma_f32_16x16x32_bf16(false, a.v, false, bm.v,
                                                       (short)0, acc[t], false, false);
    }
  }

  // ---- store pre-BN output + fused BN2 partial stats ----------------------
#pragma unroll
  for (int t = 0; t < 4; ++t) {
    int ky = t >> 1, kx = t & 1;
    int oy = 2 * iy + ky, ox = 2 * ix + kx;
#pragma unroll
    for (int r = 0; r < 8; ++r) {
      int co = cob + wv * 16 + r + 8 * hi;
      y[(((long)bb * C_ + co) * 112 + oy) * 112 + ox] = acc[t][r];
    }
  }
#pragma unroll
  for (int r = 0; r < 8; ++r) {
    int co = cob + wv * 16 + r + 8 * hi;
    float s = acc[0][r] + acc[1][r] + acc[2][r] + acc[3][r];
    float q = acc[0][r] * acc[0][r] + acc[1][r] * acc[1][r] +
              acc[2][r] * acc[2][r] + acc[3][r] * acc[3][r];
#pragma unroll
    for (int off = 1; off < 16; off <<= 1) {
      s += __shfl_xor(s, off, 32);
      q += __shfl_xor(q, off, 32);
    }
    if (mn == 0) { atomicAdd(&sum2[co], s); atomicAdd(&sq2[co], q); }
  }
}

// BN2 + ReLU in place on output.
__global__ __launch_bounds__(256) void bn_relu(
    float* __restrict__ y, const float* __restrict__ a,
    const float* __restrict__ sh, long total) {
  long i = (long)blockIdx.x * 256 + threadIdx.x;
  if (i >= total) return;
  int c = (int)((i / 12544) & (C_ - 1));
  y[i] = fmaxf(a[c] * y[i] + sh[c], 0.0f);
}

// ---------------------------------------------------------------------------
extern "C" void kernel_launch(void* const* d_in, const int* in_sizes, int n_in,
                              void* d_out, int out_size, void* d_ws, size_t ws_size,
                              hipStream_t stream) {
  (void)in_sizes; (void)n_in; (void)out_size; (void)ws_size;
  const float* x       = (const float*)d_in[0];
  const float* noise_u = (const float*)d_in[1];
  const float* w1      = (const float*)d_in[2];
  const float* b1      = (const float*)d_in[3];
  const float* g1      = (const float*)d_in[4];
  const float* beta1   = (const float*)d_in[5];
  const float* w4      = (const float*)d_in[6];
  const float* b4      = (const float*)d_in[7];
  const float* g2      = (const float*)d_in[8];
  const float* beta2   = (const float*)d_in[9];
  float* out = (float*)d_out;

  // workspace layout (floats): z1 | stats(1536) | bf16 weights
  float* wsf  = (float*)d_ws;
  float* z1   = wsf;
  float* sum1 = wsf + Z1_ELEMS;
  float* sq1  = sum1 + 128;
  float* a1   = sq1 + 128;
  float* c1   = a1 + 128;
  float* sum2 = c1 + 128;
  float* sq2  = sum2 + 256;
  float* a2   = sq2 + 256;
  float* c2   = a2 + 256;
  ushort* w1bf = (ushort*)(c2 + 256);
  ushort* w4bf = w1bf + W1BF_ELEMS;

  zero_stats<<<1, 256, 0, stream>>>(sum1);
  cvt_weights<<<(W1BF_ELEMS + 255) / 256, 256, 0, stream>>>(w1, w4, w1bf, w4bf);
  conv1_wmma<<<NTOT / 16, 256, 0, stream>>>(x, noise_u, w1bf, b1, z1, sum1, sq1);
  bn_finalize<<<1, 128, 0, stream>>>(sum1, sq1, g1, beta1, a1, c1, 128,
                                     1.0f / (float)NTOT);
  bn_sig_bec<<<(Z1_ELEMS + 255) / 256, 256, 0, stream>>>(z1, a1, c1, noise_u, Z1_ELEMS);
  deconv_wmma<<<dim3(NTOT / 16, 2), 256, (4 * 128 * 128 + 16 * 32) * sizeof(ushort),
                stream>>>(z1, w4bf, b4, out, sum2, sq2);
  bn_finalize<<<1, 256, 0, stream>>>(sum2, sq2, g2, beta2, a2, c2, 256,
                                     1.0f / (4.0f * (float)NTOT));
  bn_relu<<<(int)((OUT_ELEMS + 255) / 256), 256, 0, stream>>>(out, a2, c2, OUT_ELEMS);
}
